// LCNNConv2d_38757784879558
// MI455X (gfx1250) — matrix operand order repaired and checked by reading, not compiled, not run on hardware
//
#include <hip/hip_runtime.h>

typedef __attribute__((ext_vector_type(16))) __bf16 v16bf;
typedef __attribute__((ext_vector_type(8)))  float  v8f;

#define CIN   64
#define HW    96
#define IMG   (HW*HW)          // 9216 pixels per image plane
#define OCH   256
#define KTOT  (CIN*9)          // 576
#define BM    128
#define BN    256
#define BK    32
#define LDK   40               // padded LDS row stride in bf16 elems (80B, conflict-free)
#define NSTEP (KTOT/BK)        // 18

// ---------------------------------------------------------------------------
// Kernel 1: fold the COO lookup table + dictionary into an effective bf16
// filter:  weff[o][k],  k = p*64 + c   (p = kh*3+kw, position-major).
//   weff[o, p*64+c] = sum_e coeff[o,e] * dict[idx[o,e], c, p]
// ---------------------------------------------------------------------------
__global__ __launch_bounds__(KTOT) void build_weff_kernel(
    const float* __restrict__ dict,      // [100,64,3,3]
    const float* __restrict__ coeff,     // [256,3,3,3]
    const int*   __restrict__ idx,       // [256,3,3,3]
    void* __restrict__ weff_raw)         // [256,576] bf16
{
  __bf16* weff = (__bf16*)weff_raw;
  const int o = blockIdx.x;
  __shared__ float s_c[27];
  __shared__ int   s_i[27];
  const int t = threadIdx.x;             // 0..575
  if (t < 27) { s_c[t] = coeff[o*27 + t]; s_i[t] = idx[o*27 + t]; }
  __syncthreads();
  const int p = t / CIN;                 // kernel tap 0..8
  const int c = t - p*CIN;               // input channel
  float acc = 0.f;
  for (int e = 0; e < 27; ++e)
    acc += s_c[e] * dict[(s_i[e]*CIN + c)*9 + p];
  weff[(size_t)o*KTOT + t] = (__bf16)acc;
}

// ---------------------------------------------------------------------------
// Kernel 2: implicit-GEMM 3x3 conv via bf16 WMMA, f32 accumulate.
//   M = 256 out channels, N = 147456 pixels, K = 576 (tap-major)
// Workgroup: 256 threads (8 wave32) tiled 2(M)x4(N); block tile 128x256,
// wave tile 64x64, K-step 32, LDS double-buffer.
// Per wave / K-step: 16 ds_load_b128 -> 16 WMMA (4 per dscnt wait).
// ---------------------------------------------------------------------------
__global__ __launch_bounds__(256) void lcnn_wmma_kernel(
    const float* __restrict__ x,         // [16,64,96,96] f32
    const void*  __restrict__ weff_raw,  // [256,576] bf16
    float* __restrict__ out)             // [16,256,96,96] f32
{
  const __bf16* __restrict__ weff = (const __bf16*)weff_raw;
  __shared__ alignas(16) __bf16 sA[2][BM*LDK];   // weights tile  [m][k]
  __shared__ alignas(16) __bf16 sB[2][BN*LDK];   // im2col tile   [n][k]

  const int t    = threadIdx.x;
  const int lane = t & 31;
  const int wv   = t >> 5;               // wave id 0..7
  const int wvM  = wv >> 2;              // 0..1 -> M half (64 rows)
  const int wvN  = wv & 3;               // 0..3 -> N quarter (64 cols)
  const int m0   = blockIdx.y * BM;
  const int n0   = blockIdx.x * BN;

  // ---- hoisted staging coordinates ----
  const int arow  = t >> 1;              // A: 2 threads/row, 16 bf16 each
  const int ahalf = t & 1;
  const __bf16* gA = weff + (size_t)(m0 + arow)*KTOT + ahalf*16;
  const int nl    = t;                   // B: 1 thread/pixel, 32 channels
  const int npix  = n0 + nl;
  const int bimg  = npix / IMG;
  const int hwr   = npix - bimg*IMG;
  const int hh    = hwr / HW;
  const int ww    = hwr - hh*HW;

  v8f acc[4][4] = {};

  auto stage = [&](int buf, int kk) {
    // A tile: 128 rows x 32 k of bf16 weights (k-contiguous in global)
    {
      const uint4* g = (const uint4*)(gA + kk*BK);
      uint4 q0 = g[0], q1 = g[1];
      uint4* d = (uint4*)&sA[buf][arow*LDK + ahalf*16];
      d[0] = q0; d[1] = q1;
    }
    // B tile: on-the-fly im2col + f32->bf16; tap p fixed within a K-step.
    // Pack 8 bf16 at a time -> 4x ds_store_b128 per thread.
    {
      const int p   = kk >> 1;
      const int c0  = (kk & 1) << 5;
      const int y   = hh + p/3 - 1;
      const int xx  = ww + p - (p/3)*3 - 1;
      const bool ok = ((unsigned)y < HW) & ((unsigned)xx < HW);
      const float* src = x + ((size_t)(bimg*CIN + c0))*IMG + y*HW + xx;
      #pragma unroll
      for (int g8 = 0; g8 < 4; ++g8) {
        union { __bf16 h[8]; uint4 q; } pk;
        #pragma unroll
        for (int j = 0; j < 8; ++j) {
          float v = ok ? src[(size_t)(g8*8 + j)*IMG] : 0.f;  // zero-pad halo
          pk.h[j] = (__bf16)v;
        }
        *(uint4*)&sB[buf][nl*LDK + g8*8] = pk.q;
      }
    }
  };

  auto compute = [&](int buf) {
    union Frag { v16bf v; uint4 q[2]; };
    const int r16 = lane & 15;
    const int hi  = lane >> 4;
    // B fragments (contiguous k order, same as A): lane holds N=lane%16, K=(lane/16)*16..+15.
    // Keep all 4 live across the whole K-step.
    Frag fb[4];
    #pragma unroll
    for (int nt = 0; nt < 4; ++nt) {
      const uint4* bs =
          (const uint4*)&sB[buf][(wvN*64 + nt*16 + r16)*LDK + (hi << 4)];
      fb[nt].q[0] = bs[0]; fb[nt].q[1] = bs[1];
    }
    // Per mt: load one A fragment, then 4 back-to-back WMMAs against it.
    #pragma unroll
    for (int mt = 0; mt < 4; ++mt) {
      Frag fa;   // lane holds M=lane%16, K=(lane/16)*16..+15
      const __bf16* ap = &sA[buf][(wvM*64 + mt*16 + r16)*LDK + (hi << 4)];
      fa.q[0] = *(const uint4*)ap;
      fa.q[1] = *(const uint4*)(ap + 8);
      #pragma unroll
      for (int nt = 0; nt < 4; ++nt)
        acc[mt][nt] = __builtin_amdgcn_wmma_f32_16x16x32_bf16(
            /*neg_a=*/false, fa.v, /*neg_b=*/false, fb[nt].v,
            /*c_mod=*/(short)0, acc[mt][nt],
            /*reuse_a=*/false, /*reuse_b=*/false);
    }
  };

  stage(0, 0);
  __syncthreads();
  for (int kk = 0; kk < NSTEP; ++kk) {
    if (kk + 1 < NSTEP) stage((kk + 1) & 1, kk + 1);   // fill other buffer
    compute(kk & 1);
    __syncthreads();
  }

  // ---- epilogue: C/D layout -> NCHW f32 stores ----
  const int r16 = lane & 15;
  const int mb  = m0 + wvM*64 + ((lane >> 4) << 3);
  #pragma unroll
  for (int nt = 0; nt < 4; ++nt) {
    const int epN = n0 + wvN*64 + nt*16 + r16;
    const int eb  = epN / IMG;
    const int ehw = epN - eb*IMG;
    float* op = out + (size_t)eb*OCH*IMG + ehw;
    #pragma unroll
    for (int mt = 0; mt < 4; ++mt)
      #pragma unroll
      for (int r = 0; r < 8; ++r)
        op[(size_t)(mb + mt*16 + r)*IMG] = acc[mt][nt][r];
  }
}

// ---------------------------------------------------------------------------
extern "C" void kernel_launch(void* const* d_in, const int* in_sizes, int n_in,
                              void* d_out, int out_size, void* d_ws, size_t ws_size,
                              hipStream_t stream) {
  const float* x     = (const float*)d_in[0];   // [16,64,96,96]
  const float* dict  = (const float*)d_in[1];   // [100,64,3,3]
  const float* coeff = (const float*)d_in[2];   // [256,3,3,3]
  const int*   idx   = (const int*)d_in[3];     // [256,3,3,3]
  float* out = (float*)d_out;                   // [16,256,96,96]
  void* weff = d_ws;                            // 256*576 bf16 = 288 KB scratch

  build_weff_kernel<<<dim3(OCH), dim3(KTOT), 0, stream>>>(dict, coeff, idx, weff);
  lcnn_wmma_kernel<<<dim3((16*IMG)/BN, OCH/BM), dim3(256), 0, stream>>>(x, weff, out);
}
